// CircuitGraphModel_50921132261400
// MI455X (gfx1250) — compile-verified, matrix operand order checked by reading, not run on hardware
//
#include <hip/hip_runtime.h>

// =====================================================================
// CircuitGraphModel for MI455X (gfx1250), wave32 + WMMA bf16.
//
// Input-order assumption: harness flattens setup_inputs() as a JAX pytree
// (dicts sorted alphabetically, recursively):
//   adj, params{mlp{b1,b2,w1,w2}, p1{att_b,att_w,le1_b,le1_w,le2_w,le3_b,
//   le3_w,lin_b,lin_w}, p2{...}, t1{bk,bq,bs,bv,wk,wq,ws,wv}, t2{...}}, x
// Workspace assumption: ws_size >= ~119 MiB.
//
// All GEMM operands: A row-major [M,K], B transposed [N,K]; K % 32 == 0
// (zero-padded), so fragment loads are branch-free pairs of b128 loads.
// Masks are stored transposed+encoded in bf16: 1.0 real edge, 0.5 self-only,
// 0.0 none. Attention threshold 0.75, pooling (mask|I) threshold 0.25.
// All masked-exp paths are branchless (flag-multiply + clamped exponent).
// =====================================================================

typedef __bf16 bf16_t;
typedef __attribute__((ext_vector_type(16))) __bf16 v16bf;
typedef __attribute__((ext_vector_type(8)))  __bf16 v8bf;
typedef __attribute__((ext_vector_type(8)))  float  v8f;
typedef __attribute__((ext_vector_type(8)))  float  v8fl;

#define DEV __device__ __forceinline__

// 16-bit WMMA fragment: lane's elements = K runs [kh..kh+7] and [16+kh..16+kh+7]
DEV v16bf load_frag(const bf16_t* __restrict__ row, int lane) {
  const int kh = (lane & 16) ? 8 : 0;
  v8bf lo = *reinterpret_cast<const v8bf*>(row + kh);
  v8bf hi = *reinterpret_cast<const v8bf*>(row + 16 + kh);
  return __builtin_shufflevector(lo, hi, 0, 1, 2, 3, 4, 5, 6, 7, 8, 9, 10, 11,
                                 12, 13, 14, 15);
}

DEV v8f wmma_bf16(v16bf a, v16bf b, v8f c) {
  return __builtin_amdgcn_wmma_f32_16x16x32_bf16(false, a, false, b, (short)0, c,
                                                 false, false);
}

DEV float wred_max16(float v) {
#pragma unroll
  for (int m = 1; m < 16; m <<= 1) v = fmaxf(v, __shfl_xor(v, m, 32));
  return v;
}
DEV float wred_sum16(float v) {
#pragma unroll
  for (int m = 1; m < 16; m <<= 1) v += __shfl_xor(v, m, 32);
  return v;
}
DEV float wred_sum32(float v) {
#pragma unroll
  for (int m = 1; m < 32; m <<= 1) v += __shfl_xor(v, m, 32);
  return v;
}

DEV float lrelu02(float t) { return fmaxf(t, 0.2f * t); }  // slope 0.2, exact

// ----------------------------------------------------------------
// conversions / padding / transposes
// ----------------------------------------------------------------
__global__ void cvt_f32_bf16_kernel(const float* __restrict__ s,
                                    bf16_t* __restrict__ d, long n) {
  long i = (long)blockIdx.x * blockDim.x + threadIdx.x;
  if (i < n) d[i] = (bf16_t)s[i];
}

// dst[r, c<K ? src : 0] with row padded K -> Kpad
__global__ void pad_rows_kernel(const float* __restrict__ s, bf16_t* __restrict__ d,
                                long rows, int K, int Kpad) {
  long i = (long)blockIdx.x * blockDim.x + threadIdx.x;
  if (i >= rows * Kpad) return;
  long r = i / Kpad;
  int c = (int)(i - r * Kpad);
  d[i] = (c < K) ? (bf16_t)s[r * K + c] : (bf16_t)0.f;
}

// weights: src [K, N] row-major -> dst [N, Kpad] (transposed, zero padded)
__global__ void transpose_pad_w_kernel(const float* __restrict__ s,
                                       bf16_t* __restrict__ d, int K, int N,
                                       int Kpad) {
  long i = (long)blockIdx.x * blockDim.x + threadIdx.x;
  if (i >= (long)N * Kpad) return;
  int o = (int)(i / Kpad), kk = (int)(i % Kpad);
  d[i] = (kk < K) ? (bf16_t)s[(long)kk * N + o] : (bf16_t)0.f;
}

// activations: src [Bn*n, d] f32 -> dst [d, ldT] bf16 (column = global row)
__global__ void transpose_cvt_kernel(const float* __restrict__ s,
                                     bf16_t* __restrict__ d, int n, int dd,
                                     long ldT) {
  const int j = blockIdx.x, b = blockIdx.y, c = threadIdx.x;
  const long row = (long)b * n + j;
  d[(long)c * ldT + row] = (bf16_t)s[row * dd + c];
}

// ----------------------------------------------------------------
// Generic batched WMMA GEMM: C[M,N] = A[M,K] * Bt[N,K]^T (+bias[N])
// One wave per 16x16 tile; K % 32 == 0; CT: store C transposed [N,M].
// ----------------------------------------------------------------
template <bool CT, typename OUTT>
__global__ void gemm_kernel(const bf16_t* __restrict__ A,
                            const bf16_t* __restrict__ Bt,
                            const float* __restrict__ bias, OUTT* __restrict__ C,
                            int M, int N, int K, long sA, long sB, long sC) {
  const int lane = threadIdx.x;
  const int l15 = lane & 15;
  const int mt = blockIdx.x, nt = blockIdx.y, b = blockIdx.z;
  const bf16_t* Arow = A + (long)b * sA + (long)(mt * 16 + l15) * K;
  const bf16_t* Brow = Bt + (long)b * sB + (long)(nt * 16 + l15) * K;
  OUTT* Cb = C + (long)b * sC;
  v8f acc = {};
#pragma unroll 2
  for (int k0 = 0; k0 < K; k0 += 32) {
    v16bf a = load_frag(Arow + k0, lane);
    v16bf bb = load_frag(Brow + k0, lane);
    acc = wmma_bf16(a, bb, acc);
  }
  const int hf = (lane >> 4) & 1;
  const int ncol = nt * 16 + l15;
  const float badd = bias ? bias[ncol] : 0.f;
#pragma unroll
  for (int r = 0; r < 8; ++r) {
    const long row = mt * 16 + hf * 8 + r;
    if (CT)
      Cb[(long)ncol * M + row] = (OUTT)(acc[r] + badd);
    else
      Cb[row * N + ncol] = (OUTT)(acc[r] + badd);
  }
}

// ----------------------------------------------------------------
// Fused masked multi-head attention (flash style, softmax over sources j).
// Grid: (NN/16 target tiles, heads, B). Block = 32 (one wave).
// out pre-filled with the skip connection (x@ws+bs); we +=.
// maskT[b,i,j]: 1.0 real edge, 0.5 self-only (ignored here), 0 none.
// vT: [D_total rows, ldV cols], row = channel, col = global node.
// ----------------------------------------------------------------
template <int NN>
__global__ void attn_kernel(const bf16_t* __restrict__ q,
                            const bf16_t* __restrict__ k,
                            const bf16_t* __restrict__ vT,
                            const bf16_t* __restrict__ maskT,
                            float* __restrict__ out, int D, long ldV) {
  const int lane = threadIdx.x;
  const int l15 = lane & 15;
  const int hf = (lane >> 4) & 1;
  const int it = blockIdx.x, h = blockIdx.y, b = blockIdx.z;
  const long nb = (long)b * NN;
  const int hc0 = h * 64;

  const bf16_t* qp = q + (nb + it * 16 + l15) * D + hc0;
  v16bf qA0 = load_frag(qp, lane);
  v16bf qA1 = load_frag(qp + 32, lane);

  v8f O[4] = {{}, {}, {}, {}};
  float run_m[8], run_l[8];
#pragma unroll
  for (int r = 0; r < 8; ++r) { run_m[r] = -3.0e38f; run_l[r] = 0.f; }

  __shared__ __align__(16) bf16_t Pl[16 * 32];  // P tile (C-layout -> A-layout)

  for (int j0 = 0; j0 < NN; j0 += 32) {
    // S = Q K^T : 16 targets x 32 sources, K = 64 channels
    v8f S0 = {}, S1 = {};
    {
      const bf16_t* kp0 = k + (nb + j0 + l15) * D + hc0;
      const bf16_t* kp1 = kp0 + 16 * D;
      v16bf k0a = load_frag(kp0, lane), k0b = load_frag(kp0 + 32, lane);
      v16bf k1a = load_frag(kp1, lane), k1b = load_frag(kp1 + 32, lane);
      S0 = wmma_bf16(qA0, k0a, S0);
      S0 = wmma_bf16(qA1, k0b, S0);
      S1 = wmma_bf16(qA0, k1a, S1);
      S1 = wmma_bf16(qA1, k1b, S1);
    }
    // mask + scale + online softmax (rows live in 16-lane halves), branchless
#pragma unroll
    for (int r = 0; r < 8; ++r) {
      const int irow = it * 16 + hf * 8 + r;
      const bf16_t* mrow = maskT + (nb + irow) * (long)NN + j0;
      float f0 = ((float)mrow[l15] > 0.75f) ? 1.f : 0.f;
      float f1 = ((float)mrow[16 + l15] > 0.75f) ? 1.f : 0.f;
      float s0 = f0 > 0.f ? S0[r] * 0.125f : -1.0e30f;  // 1/sqrt(64)
      float s1 = f1 > 0.f ? S1[r] * 0.125f : -1.0e30f;
      float tm = wred_max16(fmaxf(s0, s1));
      float nm = fmaxf(run_m[r], tm);
      float corr = __expf(run_m[r] - nm);
      // exponent clamped to <=0: exact for edges, avoids inf*0 for non-edges
      float p0 = __expf(fminf(s0 - nm, 0.f)) * f0;
      float p1 = __expf(fminf(s1 - nm, 0.f)) * f1;
      run_l[r] = run_l[r] * corr + wred_sum16(p0 + p1);
      run_m[r] = nm;
#pragma unroll
      for (int ct = 0; ct < 4; ++ct) O[ct][r] *= corr;
      Pl[(hf * 8 + r) * 32 + l15] = (bf16_t)p0;
      Pl[(hf * 8 + r) * 32 + 16 + l15] = (bf16_t)p1;
    }
    // O += P (16x32) * V (32x64), 4 column tiles
    v16bf pA = load_frag(Pl + l15 * 32, lane);
#pragma unroll
    for (int ct = 0; ct < 4; ++ct) {
      v16bf vB = load_frag(vT + (long)(hc0 + ct * 16 + l15) * ldV + nb + j0, lane);
      O[ct] = wmma_bf16(pA, vB, O[ct]);
    }
  }
#pragma unroll
  for (int r = 0; r < 8; ++r) {
    const int irow = it * 16 + hf * 8 + r;
    float l = run_l[r];
    float inv = (l > 0.f) ? 1.f / l : 0.f;  // all-masked target -> 0
    float* op = out + (nb + irow) * D + hc0;
#pragma unroll
    for (int ct = 0; ct < 4; ++ct) op[ct * 16 + l15] += O[ct][r] * inv;
  }
}

// ----------------------------------------------------------------
// masked segment-max over sources (mask|I): out[b,i,c] = max_j x[b,j,c]
// (branchy on purpose: mask row is uniform across the block, skips ~98% rows)
// ----------------------------------------------------------------
__global__ void masked_max_kernel(const float* __restrict__ x,
                                  const bf16_t* __restrict__ maskT,
                                  float* __restrict__ out, int n, int d) {
  const int i = blockIdx.x, b = blockIdx.y, c = threadIdx.x;
  const long nb = (long)b * n;
  const bf16_t* mrow = maskT + (nb + i) * (long)n;
  float m = -1.0e30f;
  for (int j = 0; j < n; ++j) {
    if ((float)mrow[j] > 0.25f) m = fmaxf(m, x[(nb + j) * d + c]);
  }
  out[(nb + i) * d + c] = m;
}

// out[r] = dot(A[r,:], w) (+ bias[0]); one wave per row
__global__ void rowdot_kernel(const float* __restrict__ A,
                              const float* __restrict__ w,
                              const float* __restrict__ bias,
                              float* __restrict__ out, int d) {
  const long r = blockIdx.x;
  const int lane = threadIdx.x;
  float s = 0.f;
  for (int c = lane; c < d; c += 32) s += A[r * d + c] * w[c];
  s = wred_sum32(s);
  if (lane == 0) out[r] = s + (bias ? bias[0] : 0.f);
}

// per-target softmax stats over sources j of leaky_relu(ai+aj), mask|I
__global__ void score_stats_kernel(const float* __restrict__ ai,
                                   const float* __restrict__ aj,
                                   const bf16_t* __restrict__ maskT,
                                   float* __restrict__ mout,
                                   float* __restrict__ sinv, int n) {
  const int i = blockIdx.x, b = blockIdx.y, lane = threadIdx.x;
  const long nb = (long)b * n;
  const bf16_t* mrow = maskT + (nb + i) * (long)n;
  const float aiv = ai[nb + i];
  float mmax = -3.0e38f;
  for (int j = lane; j < n; j += 32) {
    bool e = (float)mrow[j] > 0.25f;
    float t = lrelu02(aiv + aj[nb + j]);
    mmax = fmaxf(mmax, e ? t : -3.0e38f);
  }
#pragma unroll
  for (int m = 1; m < 32; m <<= 1) mmax = fmaxf(mmax, __shfl_xor(mmax, m, 32));
  float ss = 0.f;
  for (int j = lane; j < n; j += 32) {
    float f = ((float)mrow[j] > 0.25f) ? 1.f : 0.f;
    float t = lrelu02(aiv + aj[nb + j]);
    ss += f * __expf(fminf(t - mmax, 0.f));
  }
  ss = wred_sum32(ss);
  if (lane == 0) {
    mout[nb + i] = mmax;
    sinv[nb + i] = (ss > 0.f) ? 1.f / ss : 0.f;
  }
}

// xc[b,i,c] = sum_j score[b,j,i] * x[b,j,c]; score generated in A layout.
// hT: transposed features [d rows, ldH cols]. mask fragment loaded with the
// same K-pattern as the A fragment (contiguous b128 pairs). Branchless score.
__global__ void xc_kernel(const float* __restrict__ ai, const float* __restrict__ aj,
                          const float* __restrict__ mmax,
                          const float* __restrict__ sinv,
                          const bf16_t* __restrict__ maskT,
                          const bf16_t* __restrict__ hT,
                          float* __restrict__ xcout, int n, int d, long ldH) {
  const int lane = threadIdx.x;
  const int l15 = lane & 15;
  const int kh = (lane & 16) ? 8 : 0;
  const int it = blockIdx.x, ct = blockIdx.y, b = blockIdx.z;
  const int irow = it * 16 + l15;
  const long nb = (long)b * n;
  const float aiv = ai[nb + irow];
  const float mv = mmax[nb + irow];
  const float sv = sinv[nb + irow];
  const bf16_t* mrow = maskT + (nb + irow) * (long)n;
  const bf16_t* hrow = hT + (long)(ct * 16 + l15) * ldH + nb;
  v8f acc = {};
  for (int j0 = 0; j0 < n; j0 += 32) {
    v16bf mfr = load_frag(mrow + j0, lane);
    v8fl ajlo = *reinterpret_cast<const v8fl*>(aj + nb + j0 + kh);
    v8fl ajhi = *reinterpret_cast<const v8fl*>(aj + nb + j0 + 16 + kh);
    v16bf A;
#pragma unroll
    for (int e = 0; e < 16; ++e) {
      float f = ((float)mfr[e] > 0.25f) ? sv : 0.f;
      float t = lrelu02(aiv + ((e < 8) ? ajlo[e] : ajhi[e - 8]));
      A[e] = (bf16_t)(__expf(fminf(t - mv, 0.f)) * f);
    }
    v16bf Bf = load_frag(hrow + j0, lane);
    acc = wmma_bf16(A, Bf, acc);
  }
  const int hf = (lane >> 4) & 1;
#pragma unroll
  for (int r = 0; r < 8; ++r) {
    int row = it * 16 + hf * 8 + r;
    xcout[(nb + row) * d + ct * 16 + l15] = acc[r];
  }
}

// LEConv fitness: sigmoid( sum_j mf[j,i]*a[j] - deg_i*bv[i] + cv[i] ), mask|I
__global__ void fit_kernel(const float* __restrict__ av, const float* __restrict__ bv,
                           const float* __restrict__ cv,
                           const bf16_t* __restrict__ maskT,
                           float* __restrict__ fit, int n) {
  const int i = blockIdx.x, b = blockIdx.y, lane = threadIdx.x;
  const long nb = (long)b * n;
  const bf16_t* mrow = maskT + (nb + i) * (long)n;
  float sa = 0.f, deg = 0.f;
  for (int j = lane; j < n; j += 32) {
    float f = ((float)mrow[j] > 0.25f) ? 1.f : 0.f;
    sa += f * av[nb + j];
    deg += f;
  }
  sa = wred_sum32(sa);
  deg = wred_sum32(deg);
  if (lane == 0) {
    float z = sa - deg * bv[nb + i] + cv[nb + i];
    fit[nb + i] = 1.f / (1.f + __expf(-z));
  }
}

// per-graph top-k selection (descending, low-index tie-break)
__global__ void topk_kernel(const float* __restrict__ fit, int* __restrict__ perm,
                            float* __restrict__ fv, int n, int kk) {
  const int b = blockIdx.x, lane = threadIdx.x;
  __shared__ float vals[1024];
  for (int j = lane; j < n; j += 32) vals[j] = fit[(long)b * n + j];
  __syncthreads();
  for (int t = 0; t < kk; ++t) {
    float best = -3.0e38f;
    int bi = 1 << 30;
    for (int j = lane; j < n; j += 32) {
      float v = vals[j];
      if (v > best) { best = v; bi = j; }
    }
#pragma unroll
    for (int m = 1; m < 32; m <<= 1) {
      float ov = __shfl_xor(best, m, 32);
      int oi = __shfl_xor(bi, m, 32);
      if (ov > best || (ov == best && oi < bi)) { best = ov; bi = oi; }
    }
    if (lane == 0) {
      perm[b * kk + t] = bi;
      fv[b * kk + t] = best;
      vals[bi] = -3.0e38f;
    }
    __syncthreads();
  }
}

__global__ void gather_scale_kernel(const float* __restrict__ xc,
                                    const int* __restrict__ perm,
                                    const float* __restrict__ fv,
                                    float* __restrict__ out, int n, int kk, int d) {
  const int p = blockIdx.x, b = blockIdx.y, c = threadIdx.x;
  const int pi = perm[b * kk + p];
  const float f = fv[b * kk + p];
  out[((long)b * kk + p) * d + c] = xc[((long)b * n + pi) * d + c] * f;
}

// encoded transposed mask from adj: mT[b,i,j] = edge(j->i)?1 : (i==j?0.5:0)
__global__ void build_mt_kernel(const float* __restrict__ adj,
                                bf16_t* __restrict__ mt, int n) {
  const int i = blockIdx.x, b = blockIdx.y;
  const long nb = (long)b * n;
  for (int j = threadIdx.x; j < n; j += blockDim.x) {
    float v = (adj[(nb + j) * n + i] > 0.f) ? 1.f : (i == j ? 0.5f : 0.f);
    mt[(nb + i) * n + j] = (bf16_t)v;
  }
}

// mf[b,j,l] = (adj>0 || l==j) as exact 0/1 (GEMM A operand, [j,l] row-major)
__global__ void build_mf_kernel(const float* __restrict__ adj,
                                bf16_t* __restrict__ mf, int n) {
  const int j = blockIdx.x, b = blockIdx.y;
  const long base = ((long)b * n + j) * n;
  for (int i = threadIdx.x; i < n; i += blockDim.x)
    mf[base + i] = (adj[base + i] > 0.f || i == j) ? (bf16_t)1.f : (bf16_t)0.f;
}

// SpT[b,q,j] = mf[b,j,perm[q]]  (transposed cluster-assignment columns)
__global__ void build_spt_kernel(const float* __restrict__ adj,
                                 const int* __restrict__ perm,
                                 bf16_t* __restrict__ SpT, int n, int kk) {
  const int qq = blockIdx.x, b = blockIdx.y;
  const int pi = perm[b * kk + qq];
  const long nb = (long)b * n;
  for (int j = threadIdx.x; j < n; j += blockDim.x) {
    bool e = (adj[(nb + j) * n + pi] > 0.f) || (j == pi);
    SpT[((long)b * kk + qq) * n + j] = e ? (bf16_t)1.f : (bf16_t)0.f;
  }
}

// mask1T[b,i,j] = (M2[j,i]>0 && j!=i)?1 : (i==j?0.5:0)
__global__ void build_mask1t_kernel(const float* __restrict__ M2,
                                    bf16_t* __restrict__ mt, int kk) {
  const int i = blockIdx.x, b = blockIdx.y;
  const long kb = (long)b * kk;
  for (int j = threadIdx.x; j < kk; j += blockDim.x) {
    bool real = (M2[(kb + j) * kk + i] > 0.5f) && (i != j);
    float v = real ? 1.f : (i == j ? 0.5f : 0.f);
    mt[(kb + i) * kk + j] = (bf16_t)v;
  }
}

__global__ void mean_kernel(const float* __restrict__ x, float* __restrict__ g,
                            int kk, int d) {
  const int b = blockIdx.x, c = threadIdx.x;
  float s = 0.f;
  for (int p = 0; p < kk; ++p) s += x[((long)b * kk + p) * d + c];
  g[b * d + c] = s / (float)kk;
}

__global__ void mlp_kernel(const float* __restrict__ g, const float* __restrict__ w1,
                           const float* __restrict__ b1, const float* __restrict__ w2,
                           const float* __restrict__ b2, float* __restrict__ out) {
  const int b = blockIdx.x, t = threadIdx.x;  // 64 threads
  __shared__ float hbuf[64];
  float s = b1[t];
  for (int c = 0; c < 128; ++c) s += g[b * 128 + c] * w1[c * 64 + t];
  hbuf[t] = s;
  __syncthreads();
  if (t < 4) {
    float o = b2[t];
    for (int c = 0; c < 64; ++c) o += hbuf[c] * w2[c * 4 + t];
    out[b * 4 + t] = o;
  }
}

// =====================================================================
// Host side
// =====================================================================
enum {
  IN_ADJ = 0,
  IN_MLP_B1, IN_MLP_B2, IN_MLP_W1, IN_MLP_W2,
  IN_P1_ATT_B, IN_P1_ATT_W, IN_P1_LE1_B, IN_P1_LE1_W, IN_P1_LE2_W,
  IN_P1_LE3_B, IN_P1_LE3_W, IN_P1_LIN_B, IN_P1_LIN_W,
  IN_P2_ATT_B, IN_P2_ATT_W, IN_P2_LE1_B, IN_P2_LE1_W, IN_P2_LE2_W,
  IN_P2_LE3_B, IN_P2_LE3_W, IN_P2_LIN_B, IN_P2_LIN_W,
  IN_T1_BK, IN_T1_BQ, IN_T1_BS, IN_T1_BV, IN_T1_WK, IN_T1_WQ, IN_T1_WS, IN_T1_WV,
  IN_T2_BK, IN_T2_BQ, IN_T2_BS, IN_T2_BV, IN_T2_WK, IN_T2_WQ, IN_T2_WS, IN_T2_WV,
  IN_X
};

extern "C" void kernel_launch(void* const* d_in, const int* in_sizes, int n_in,
                              void* d_out, int out_size, void* d_ws, size_t ws_size,
                              hipStream_t stream) {
  (void)in_sizes; (void)n_in; (void)out_size; (void)ws_size;
  const int Bn = 8, N1 = 1024, N2 = 512, Fin = 16, FinP = 32;
  const int D1 = 192, D2 = 128, K1 = 512, K2 = 256;
  const long R1 = (long)Bn * N1;  // 8192 global nodes stage 1
  const long R2 = (long)Bn * N2;  // 4096 global nodes stage 2

  const float* adj = (const float*)d_in[IN_ADJ];
  const float* xin = (const float*)d_in[IN_X];

  // ---- workspace bump allocator ----
  char* base = (char*)d_ws;
  size_t off = 0;
  auto alloc = [&](size_t bytes) -> char* {
    off = (off + 255) & ~(size_t)255;
    char* p = base + off;
    off += bytes;
    return p;
  };
  auto allocF = [&](long n) { return (float*)alloc((size_t)n * 4); };
  auto allocB = [&](long n) { return (bf16_t*)alloc((size_t)n * 2); };
  auto allocI = [&](long n) { return (int*)alloc((size_t)n * 4); };

  bf16_t* x_bf = allocB(R1 * FinP);  // zero-padded K=16->32
  bf16_t* wq1t = allocB(D1 * FinP); bf16_t* wk1t = allocB(D1 * FinP);
  bf16_t* wv1t = allocB(D1 * FinP); bf16_t* ws1t = allocB(D1 * FinP);
  bf16_t* wq2t = allocB(D2 * D1);   bf16_t* wk2t = allocB(D2 * D1);
  bf16_t* wv2t = allocB(D2 * D1);   bf16_t* ws2t = allocB(D2 * D1);
  bf16_t* lin1t = allocB(D1 * D1);  bf16_t* lin2t = allocB(D2 * D2);

  const long E1 = R1 * D1;
  bf16_t* q1b = allocB(E1); bf16_t* k1b = allocB(E1); bf16_t* vT1 = allocB(E1);
  float* h1 = allocF(E1);   bf16_t* hT1 = allocB(E1);
  float* mx1 = allocF(E1);  bf16_t* mx1b = allocB(E1);
  float* xq1 = allocF(E1);  float* xc1 = allocF(E1);
  float* ai1 = allocF(R1); float* aj1 = allocF(R1);
  float* ms1 = allocF(R1); float* si1 = allocF(R1);
  float* av1 = allocF(R1); float* bv1 = allocF(R1);
  float* cv1 = allocF(R1); float* fit1 = allocF(R1);
  int* perm1 = allocI((long)Bn * K1); float* fv1 = allocF((long)Bn * K1);
  float* x2 = allocF((long)Bn * K1 * D1); bf16_t* x2b = allocB((long)Bn * K1 * D1);

  bf16_t* mT0 = allocB((long)Bn * N1 * N1);   // encoded transposed mask, 16 MB
  bf16_t* mf0 = allocB((long)Bn * N1 * N1);   // 0/1 [j,l] GEMM operand, 16 MB
  bf16_t* SpT = allocB((long)Bn * K1 * N1);   // 8 MB
  bf16_t* Tt = allocB((long)Bn * K1 * N1);    // 8 MB
  float* M2 = allocF((long)Bn * K1 * K1);     // 8 MB
  bf16_t* mask1T = allocB((long)Bn * K1 * K1);// 4 MB

  const long E2 = R2 * D2;
  bf16_t* q2b = allocB(E2); bf16_t* k2b = allocB(E2); bf16_t* vT2 = allocB(E2);
  float* h2 = allocF(E2);   bf16_t* hT2 = allocB(E2);
  float* mx2 = allocF(E2);  bf16_t* mx2b = allocB(E2);
  float* xq2 = allocF(E2);  float* xc2 = allocF(E2);
  float* ai2 = allocF(R2); float* aj2 = allocF(R2);
  float* ms2 = allocF(R2); float* si2 = allocF(R2);
  float* av2 = allocF(R2); float* bv2 = allocF(R2);
  float* cv2 = allocF(R2); float* fit2 = allocF(R2);
  int* perm2 = allocI((long)Bn * K2); float* fv2 = allocF((long)Bn * K2);
  float* x3 = allocF((long)Bn * K2 * D2);
  float* g = allocF((long)Bn * D2);

  auto cvt = [&](const float* s, bf16_t* d, long n) {
    cvt_f32_bf16_kernel<<<(int)((n + 255) / 256), 256, 0, stream>>>(s, d, n);
  };
  auto twpad = [&](int widx, bf16_t* d, int K, int N, int Kpad) {
    long tot = (long)N * Kpad;
    transpose_pad_w_kernel<<<(int)((tot + 255) / 256), 256, 0, stream>>>(
        (const float*)d_in[widx], d, K, N, Kpad);
  };
  auto gemmN_bf = [&](const bf16_t* A, const bf16_t* Bt, const float* bias,
                      bf16_t* C, int M, int N, int K, long sA, long sB, long sC,
                      int nb) {
    gemm_kernel<false, bf16_t><<<dim3(M / 16, N / 16, nb), 32, 0, stream>>>(
        A, Bt, bias, C, M, N, K, sA, sB, sC);
  };
  auto gemmT_bf = [&](const bf16_t* A, const bf16_t* Bt, const float* bias,
                      bf16_t* C, int M, int N, int K, long sA, long sB, long sC,
                      int nb) {
    gemm_kernel<true, bf16_t><<<dim3(M / 16, N / 16, nb), 32, 0, stream>>>(
        A, Bt, bias, C, M, N, K, sA, sB, sC);
  };
  auto gemmN_f = [&](const bf16_t* A, const bf16_t* Bt, const float* bias, float* C,
                     int M, int N, int K, long sA, long sB, long sC, int nb) {
    gemm_kernel<false, float><<<dim3(M / 16, N / 16, nb), 32, 0, stream>>>(
        A, Bt, bias, C, M, N, K, sA, sB, sC);
  };

  // ---- input padding / weight transposition ----
  pad_rows_kernel<<<(int)((R1 * FinP + 255) / 256), 256, 0, stream>>>(xin, x_bf,
                                                                      R1, Fin, FinP);
  twpad(IN_T1_WQ, wq1t, Fin, D1, FinP);
  twpad(IN_T1_WK, wk1t, Fin, D1, FinP);
  twpad(IN_T1_WV, wv1t, Fin, D1, FinP);
  twpad(IN_T1_WS, ws1t, Fin, D1, FinP);
  twpad(IN_T2_WQ, wq2t, D1, D2, D1);
  twpad(IN_T2_WK, wk2t, D1, D2, D1);
  twpad(IN_T2_WV, wv2t, D1, D2, D1);
  twpad(IN_T2_WS, ws2t, D1, D2, D1);
  twpad(IN_P1_LIN_W, lin1t, D1, D1, D1);
  twpad(IN_P2_LIN_W, lin2t, D2, D2, D2);
  build_mt_kernel<<<dim3(N1, Bn), 256, 0, stream>>>(adj, mT0, N1);

  // ================= Stage 1: TransformerConv(16 -> 192, 3 heads) =========
  gemmN_bf(x_bf, wq1t, (const float*)d_in[IN_T1_BQ], q1b, R1, D1, FinP, 0, 0, 0, 1);
  gemmN_bf(x_bf, wk1t, (const float*)d_in[IN_T1_BK], k1b, R1, D1, FinP, 0, 0, 0, 1);
  gemmT_bf(x_bf, wv1t, (const float*)d_in[IN_T1_BV], vT1, R1, D1, FinP, 0, 0, 0, 1);
  gemmN_f(x_bf, ws1t, (const float*)d_in[IN_T1_BS], h1, R1, D1, FinP, 0, 0, 0, 1);
  attn_kernel<1024><<<dim3(N1 / 16, 3, Bn), 32, 0, stream>>>(q1b, k1b, vT1, mT0,
                                                             h1, D1, R1);
  transpose_cvt_kernel<<<dim3(N1, Bn), D1, 0, stream>>>(h1, hT1, N1, D1, R1);

  // ================= Stage 1: ASAPool (1024 -> 512) ========================
  masked_max_kernel<<<dim3(N1, Bn), D1, 0, stream>>>(h1, mT0, mx1, N1, D1);
  cvt(mx1, mx1b, E1);
  gemmN_f(mx1b, lin1t, (const float*)d_in[IN_P1_LIN_B], xq1, R1, D1, D1, 0, 0, 0, 1);
  rowdot_kernel<<<R1, 32, 0, stream>>>(xq1, (const float*)d_in[IN_P1_ATT_W],
                                       (const float*)d_in[IN_P1_ATT_B], ai1, D1);
  rowdot_kernel<<<R1, 32, 0, stream>>>(h1, (const float*)d_in[IN_P1_ATT_W] + D1,
                                       nullptr, aj1, D1);
  score_stats_kernel<<<dim3(N1, Bn), 32, 0, stream>>>(ai1, aj1, mT0, ms1, si1, N1);
  xc_kernel<<<dim3(N1 / 16, D1 / 16, Bn), 32, 0, stream>>>(ai1, aj1, ms1, si1, mT0,
                                                           hT1, xc1, N1, D1, R1);
  rowdot_kernel<<<R1, 32, 0, stream>>>(xc1, (const float*)d_in[IN_P1_LE1_W],
                                       (const float*)d_in[IN_P1_LE1_B], av1, D1);
  rowdot_kernel<<<R1, 32, 0, stream>>>(xc1, (const float*)d_in[IN_P1_LE2_W],
                                       nullptr, bv1, D1);
  rowdot_kernel<<<R1, 32, 0, stream>>>(xc1, (const float*)d_in[IN_P1_LE3_W],
                                       (const float*)d_in[IN_P1_LE3_B], cv1, D1);
  fit_kernel<<<dim3(N1, Bn), 32, 0, stream>>>(av1, bv1, cv1, mT0, fit1, N1);
  topk_kernel<<<Bn, 32, 0, stream>>>(fit1, perm1, fv1, N1, K1);
  gather_scale_kernel<<<dim3(K1, Bn), D1, 0, stream>>>(xc1, perm1, fv1, x2, N1, K1, D1);

  // mask coarsening: mask1 = (Sp^T (mf Sp) > 0) & ~I   (WMMA bf16, exact 0/1)
  build_mf_kernel<<<dim3(N1, Bn), 256, 0, stream>>>(adj, mf0, N1);
  build_spt_kernel<<<dim3(K1, Bn), 256, 0, stream>>>(adj, perm1, SpT, N1, K1);
  // T[j,q] = sum_l mf[j,l] Sp[l,q]; store transposed -> Tt[q,j]
  gemmT_bf(mf0, SpT, nullptr, Tt, N1, K1, N1, (long)N1 * N1, (long)K1 * N1,
           (long)K1 * N1, Bn);
  // M2[p,q] = sum_j SpT[p,j] Tt[q,j]
  gemmN_f(SpT, Tt, nullptr, M2, K1, K1, N1, (long)K1 * N1, (long)K1 * N1,
          (long)K1 * K1, Bn);
  build_mask1t_kernel<<<dim3(K1, Bn), 256, 0, stream>>>(M2, mask1T, K1);

  // ================= Stage 2: TransformerConv(192 -> 128, 2 heads) =========
  cvt(x2, x2b, (long)Bn * K1 * D1);
  gemmN_bf(x2b, wq2t, (const float*)d_in[IN_T2_BQ], q2b, R2, D2, D1, 0, 0, 0, 1);
  gemmN_bf(x2b, wk2t, (const float*)d_in[IN_T2_BK], k2b, R2, D2, D1, 0, 0, 0, 1);
  gemmT_bf(x2b, wv2t, (const float*)d_in[IN_T2_BV], vT2, R2, D2, D1, 0, 0, 0, 1);
  gemmN_f(x2b, ws2t, (const float*)d_in[IN_T2_BS], h2, R2, D2, D1, 0, 0, 0, 1);
  attn_kernel<512><<<dim3(N2 / 16, 2, Bn), 32, 0, stream>>>(q2b, k2b, vT2, mask1T,
                                                            h2, D2, R2);
  transpose_cvt_kernel<<<dim3(N2, Bn), D2, 0, stream>>>(h2, hT2, N2, D2, R2);

  // ================= Stage 2: ASAPool (512 -> 256, mask output unused) =====
  masked_max_kernel<<<dim3(N2, Bn), D2, 0, stream>>>(h2, mask1T, mx2, N2, D2);
  cvt(mx2, mx2b, E2);
  gemmN_f(mx2b, lin2t, (const float*)d_in[IN_P2_LIN_B], xq2, R2, D2, D2, 0, 0, 0, 1);
  rowdot_kernel<<<R2, 32, 0, stream>>>(xq2, (const float*)d_in[IN_P2_ATT_W],
                                       (const float*)d_in[IN_P2_ATT_B], ai2, D2);
  rowdot_kernel<<<R2, 32, 0, stream>>>(h2, (const float*)d_in[IN_P2_ATT_W] + D2,
                                       nullptr, aj2, D2);
  score_stats_kernel<<<dim3(N2, Bn), 32, 0, stream>>>(ai2, aj2, mask1T, ms2, si2, N2);
  xc_kernel<<<dim3(N2 / 16, D2 / 16, Bn), 32, 0, stream>>>(ai2, aj2, ms2, si2,
                                                           mask1T, hT2, xc2, N2, D2,
                                                           R2);
  rowdot_kernel<<<R2, 32, 0, stream>>>(xc2, (const float*)d_in[IN_P2_LE1_W],
                                       (const float*)d_in[IN_P2_LE1_B], av2, D2);
  rowdot_kernel<<<R2, 32, 0, stream>>>(xc2, (const float*)d_in[IN_P2_LE2_W],
                                       nullptr, bv2, D2);
  rowdot_kernel<<<R2, 32, 0, stream>>>(xc2, (const float*)d_in[IN_P2_LE3_W],
                                       (const float*)d_in[IN_P2_LE3_B], cv2, D2);
  fit_kernel<<<dim3(N2, Bn), 32, 0, stream>>>(av2, bv2, cv2, mask1T, fit2, N2);
  topk_kernel<<<Bn, 32, 0, stream>>>(fit2, perm2, fv2, N2, K2);
  gather_scale_kernel<<<dim3(K2, Bn), D2, 0, stream>>>(xc2, perm2, fv2, x3, N2, K2, D2);

  // ================= readout ==============================================
  mean_kernel<<<Bn, D2, 0, stream>>>(x3, g, K2, D2);
  mlp_kernel<<<Bn, 64, 0, stream>>>(g, (const float*)d_in[IN_MLP_W1],
                                    (const float*)d_in[IN_MLP_B1],
                                    (const float*)d_in[IN_MLP_W2],
                                    (const float*)d_in[IN_MLP_B2], (float*)d_out);
}
